// LinearVC_63230508532562
// MI455X (gfx1250) — matrix operand order repaired and checked by reading, not compiled, not run on hardware
//
#include <hip/hip_runtime.h>

typedef __attribute__((ext_vector_type(16))) _Float16 v16h;
typedef __attribute__((ext_vector_type(8)))  float    v8f;

#define DIM 1024

// ---------------------------------------------------------------------------
// Kernel 1: L2-normalize each row (f32 in) and emit f16 row-major matrix.
// ---------------------------------------------------------------------------
__global__ __launch_bounds__(256)
void rownorm_f16(const float* __restrict__ in, _Float16* __restrict__ out) {
  const int row = blockIdx.x;
  const int tid = threadIdx.x;
  const float4 v = ((const float4*)(in + (size_t)row * DIM))[tid];
  float ss = v.x * v.x + v.y * v.y + v.z * v.z + v.w * v.w;
  #pragma unroll
  for (int off = 16; off > 0; off >>= 1) ss += __shfl_xor(ss, off, 32);
  __shared__ float red[8];
  if ((tid & 31) == 0) red[tid >> 5] = ss;
  __syncthreads();
  float tot = 0.0f;
  #pragma unroll
  for (int i = 0; i < 8; ++i) tot += red[i];
  const float nrm = fmaxf(sqrtf(tot), 1e-12f);
  const float s = 1.0f / nrm;
  union { _Float16 h[4]; uint2 u; } p;
  p.h[0] = (_Float16)(v.x * s);
  p.h[1] = (_Float16)(v.y * s);
  p.h[2] = (_Float16)(v.z * s);
  p.h[3] = (_Float16)(v.w * s);
  ((uint2*)(out + (size_t)row * DIM))[tid] = p.u;
}

// ---------------------------------------------------------------------------
// Kernel 2: fused similarity GEMM + row argmax.
// Block = 256 threads (8 waves); block owns 128 source rows (16 per wave).
// N in 128-column macro tiles, K in 32-element slices. A/B k-slices are
// DMA'd into LDS with GLOBAL_LOAD_ASYNC_TO_LDS_B128 (ASYNCcnt), 4-deep
// pipelined (80 KB LDS), one barrier per slice. B fragments are ping-pong
// double-buffered in registers so ds_load latency overlaps the WMMAs.
// ---------------------------------------------------------------------------
#define NBUF 4
#define KSLICES (DIM / 32)   // 32, multiple of NBUF

__global__ __launch_bounds__(256)
void simmax_kernel(const _Float16* __restrict__ Sn, const _Float16* __restrict__ Tn,
                   int* __restrict__ best, int n_tgt) {
  __shared__ uint4 ldsA[NBUF][128 * 5];   // 128 rows x 32 f16, pitch 5 uint4 (80 B)
  __shared__ uint4 ldsB[NBUF][128 * 5];   // 128 cols x 32 f16

  const int tid  = threadIdx.x;
  const int wave = tid >> 5;
  const int lane = tid & 31;
  const int hv   = lane >> 4;
  const int l16  = lane & 15;
  const int m_base = blockIdx.x * 128;
  const int aRow   = wave * 16 + l16;

  // staging assignment: thread -> (row sr, 16-element chunk sk)
  const int sr = tid >> 1;
  const int sk = (tid & 1) * 2;           // uint4 index within 32-elem row

  // 32-bit global byte offsets vs the SGPR base (GVS addressing mode)
  const unsigned aOffBase = (unsigned)(((m_base + sr) * DIM + sk * 8) * 2);
  const unsigned bRowOff  = (unsigned)((sr * DIM + sk * 8) * 2);
  const unsigned long long aBase = (unsigned long long)Sn;
  const unsigned long long bBase = (unsigned long long)Tn;

  // per-thread LDS byte addresses (low 32 bits of flat == LDS offset)
  unsigned la[NBUF], lb[NBUF];
  #pragma unroll
  for (int b = 0; b < NBUF; ++b) {
    la[b] = (unsigned)(size_t)&ldsA[b][sr * 5 + sk];
    lb[b] = (unsigned)(size_t)&ldsB[b][sr * 5 + sk];
  }

  const int nTiles = n_tgt / 128;
  const int total  = nTiles * KSLICES;    // flattened (nt, k) tile count

  // async DMA: one 128x32 A slice + one 128x32 B slice -> LDS buffer b.
  // INST_OFFSET applies to both LDS and global addresses, so offset:16
  // covers the second 16B half of each 32B chunk.
  auto issue = [&](int t, int b) {
    const int ntIdx = t >> 5;             // KSLICES == 32 slices per nt
    const int k     = (t & 31) * 32;
    const unsigned aOff = aOffBase + (unsigned)(k * 2);
    const unsigned bOff = bRowOff + (unsigned)(ntIdx * 128 * DIM * 2 + k * 2);
    asm volatile("global_load_async_to_lds_b128 %0, %1, %2"
                 :: "v"(la[b]), "v"(aOff), "s"(aBase) : "memory");
    asm volatile("global_load_async_to_lds_b128 %0, %1, %2 offset:16"
                 :: "v"(la[b]), "v"(aOff), "s"(aBase) : "memory");
    asm volatile("global_load_async_to_lds_b128 %0, %1, %2"
                 :: "v"(lb[b]), "v"(bOff), "s"(bBase) : "memory");
    asm volatile("global_load_async_to_lds_b128 %0, %1, %2 offset:16"
                 :: "v"(lb[b]), "v"(bOff), "s"(bBase) : "memory");
  };

  float run_val[8];
  int   run_idx[8];
  #pragma unroll
  for (int r = 0; r < 8; ++r) { run_val[r] = -3.402823466e+38f; run_idx[r] = 0; }

  // prologue: fill 3 pipeline stages (total >= 3)
  issue(0, 0);
  issue(1, 1);
  issue(2, 2);

  for (int ntIdx = 0; ntIdx < nTiles; ++ntIdx) {
    v8f acc[8];
    #pragma unroll
    for (int j = 0; j < 8; ++j)
      #pragma unroll
      for (int r = 0; r < 8; ++r) acc[j][r] = 0.0f;

    for (int kk = 0; kk < KSLICES; ++kk) {
      const int t = ntIdx * KSLICES + kk;
      const int b = kk & (NBUF - 1);      // ntIdx*KSLICES is a multiple of NBUF

      // tile t's 4 async ops complete (in-order); tiles t+1,t+2 stay in flight
#if __has_builtin(__builtin_amdgcn_s_wait_asynccnt)
      __builtin_amdgcn_s_wait_asynccnt(8);
#else
      asm volatile("s_wait_asynccnt 0x8" ::: "memory");
#endif
      __syncthreads();   // all waves' tile-t data visible; prior buffer reads done

      {  // prefetch tile t+3 (clamped tail issues are dead but keep counts exact)
        int tp = t + 3;
        if (tp > total - 1) tp = total - 1;
        issue(tp, tp & (NBUF - 1));
      }

      // A fragment: lane = hv*16 + m ; v0..7 <- K=hv*16..+15
      union BU { uint4 u[2]; v16h h; };
      BU A;
      A.u[0] = ldsA[b][aRow * 5 + hv * 2];
      A.u[1] = ldsA[b][aRow * 5 + hv * 2 + 1];

      // B fragments (lane = hv*16 + n ; v0..7 <- K=hv*16..+15), ping-pong
      // register double-buffered: load frag j+1 before the WMMA consuming j.
      BU Bp[2];
      Bp[0].u[0] = ldsB[b][l16 * 5 + hv * 2];
      Bp[0].u[1] = ldsB[b][l16 * 5 + hv * 2 + 1];
      #pragma unroll
      for (int j = 0; j < 8; ++j) {
        if (j < 7) {
          const int c = (j + 1) * 16 + l16;
          Bp[(j + 1) & 1].u[0] = ldsB[b][c * 5 + hv * 2];
          Bp[(j + 1) & 1].u[1] = ldsB[b][c * 5 + hv * 2 + 1];
        }
        acc[j] = __builtin_amdgcn_wmma_f32_16x16x32_f16(
            false, A.h, false, Bp[j & 1].h, (short)0, acc[j], false, false);
      }
    }

    // running argmax: reg r, lanes 0-15: (row r, N=l16); lanes 16-31: row r+8
    const int nt = ntIdx * 128;
    #pragma unroll
    for (int j = 0; j < 8; ++j) {
      const int n = nt + j * 16 + l16;
      #pragma unroll
      for (int r = 0; r < 8; ++r) {
        const float vv = acc[j][r];
        if (vv > run_val[r]) { run_val[r] = vv; run_idx[r] = n; }
      }
    }
  }

  // reduce across the 16 lanes of each half (xor stays within a half)
  #pragma unroll
  for (int off = 1; off < 16; off <<= 1) {
    #pragma unroll
    for (int r = 0; r < 8; ++r) {
      const float ov = __shfl_xor(run_val[r], off, 32);
      const int   oi = __shfl_xor(run_idx[r], off, 32);
      if (ov > run_val[r] || (ov == run_val[r] && oi < run_idx[r])) {
        run_val[r] = ov; run_idx[r] = oi;
      }
    }
  }
  if (l16 == 0) {
    const int rowBase = m_base + wave * 16 + hv * 8;
    #pragma unroll
    for (int r = 0; r < 8; ++r) best[rowBase + r] = run_idx[r];
  }
}

// ---------------------------------------------------------------------------
// Kernel 3: gather matched raw target rows (f32), one block per source row.
// ---------------------------------------------------------------------------
__global__ __launch_bounds__(256)
void gather_rows(const float* __restrict__ tgt, const int* __restrict__ idx,
                 float* __restrict__ out) {
  const int s = blockIdx.x;
  const int t = idx[s];
  const float4* src = (const float4*)(tgt + (size_t)t * DIM);
  float4* dst = (float4*)(out + (size_t)s * DIM);
  dst[threadIdx.x] = src[threadIdx.x];
}

extern "C" void kernel_launch(void* const* d_in, const int* in_sizes, int n_in,
                              void* d_out, int out_size, void* d_ws, size_t ws_size,
                              hipStream_t stream) {
  const float* src = (const float*)d_in[0];
  const float* tgt = (const float*)d_in[1];
  float* out = (float*)d_out;
  const int n_src = in_sizes[0] / DIM;   // 8192
  const int n_tgt = in_sizes[1] / DIM;   // 8192

  char* ws = (char*)d_ws;
  _Float16* Sn = (_Float16*)ws;
  _Float16* Tn = (_Float16*)(ws + (size_t)n_src * DIM * sizeof(_Float16));
  int* best = (int*)(ws + (size_t)(n_src + n_tgt) * DIM * sizeof(_Float16));

  rownorm_f16<<<n_src, 256, 0, stream>>>(src, Sn);
  rownorm_f16<<<n_tgt, 256, 0, stream>>>(tgt, Tn);
  simmax_kernel<<<n_src / 128, 256, 0, stream>>>(Sn, Tn, best, n_tgt);
  gather_rows<<<n_src, 256, 0, stream>>>(tgt, best, out);
}
